// LRPViewOfRelationNetwork_75428215652465
// MI455X (gfx1250) — matrix operand order repaired and checked by reading, not compiled, not run on hardware
//
#include <hip/hip_runtime.h>
#include <hip/hip_bf16.h>

typedef __attribute__((ext_vector_type(16))) _Float16 v16h;
typedef __attribute__((ext_vector_type(8)))  float    v8f;

#define THREADS 256

// ---------------- problem constants ----------------
#define BATCH 32
#define CH    24      // conv hidden channels
#define NPAIR 64      // 8*8 spatial positions
#define CC    26      // conv channels + 2 coord channels
#define QDIM  128     // lstm hidden
#define K0    180     // g0 input features
#define HID   256     // mlp hidden
#define NANS  29

// ---------------- workspace layout (bytes) ----------------
#define OFF_W0   0u          // 6*16*512 halves  = 98304 B
#define OFF_W1   98304u      // 8*16*512 halves  = 131072 B
#define OFF_W2   229376u
#define OFF_W3   360448u
#define OFF_CONV 491520u     // f16 [32][64][26]
#define OFF_LSTM 598016u     // f16 [32][128]
#define OFF_H0   606208u     // f32 [32][24][64][64]
#define OFF_H1   13189120u   // f32 [32][24][32][32]
#define OFF_H2   16334848u   // f32 [32][24][16][16]
#define OFF_GSUM 17121280u   // f32 [32][256]
#define OFF_FHID 17154048u   // f32 [32][256]

// ---------------- shared memory layout for rn_gmlp ----------------
#define AST     264                      // activation row stride in halves (256 + 8 pad)
#define SH_ACT  (128 * AST)              // halves per activation buffer (33792)
#define SH_W    65536                    // halves for weight buffer (max layer, 128 KB)
#define SMEM_BYTES ((2 * SH_ACT + SH_W) * 2 + 256 * 4 + 256 * 4)   // 268288 B
static_assert(SMEM_BYTES <= 320 * 1024, "LDS budget");

// =====================================================================
// Weight prep: fp32 [256][K] row-major  ->  f16 B-fragment-swizzled.
// Fragment f = kt*16+nt is 512 halves:
//   half index = blk*256 + lane*8 + h
//   k = kt*32 + blk*16 + (lane>=16 ? 8 : 0) + h ;  n = nt*16 + (lane&15)
// =====================================================================
__global__ void wprep(const float* __restrict__ w, _Float16* __restrict__ dst,
                      int K, int nhalfs) {
    int e = blockIdx.x * THREADS + threadIdx.x;
    if (e >= nhalfs) return;
    int f    = e >> 9;
    int rem  = e & 511;
    int blk  = rem >> 8;
    int lane = (rem >> 3) & 31;
    int h    = rem & 7;
    int kt = f >> 4;
    int nt = f & 15;
    int k = kt * 32 + blk * 16 + ((lane >= 16) ? 8 : 0) + h;
    int n = nt * 16 + (lane & 15);
    float v = (k < K) ? w[n * K + k] : 0.0f;
    dst[e] = (_Float16)v;
}

// =====================================================================
// Direct stride-2 3x3 conv + bias + relu (tiny share of total FLOPs)
// =====================================================================
__global__ void conv_s2_relu(const float* __restrict__ in, const float* __restrict__ w,
                             const float* __restrict__ bias, float* __restrict__ out,
                             int IC, int IH, int OH, int total) {
    int idx = blockIdx.x * THREADS + threadIdx.x;
    if (idx >= total) return;
    int ox = idx % OH;
    int oy = (idx / OH) % OH;
    int oc = (idx / (OH * OH)) % CH;
    int b  = idx / (CH * OH * OH);
    float acc = bias[oc];
    const float* wc = w + oc * IC * 9;
    for (int ic = 0; ic < IC; ++ic) {
        const float* ip = in + ((size_t)(b * IC + ic) * IH) * IH;
        const float* wp = wc + ic * 9;
        for (int ky = 0; ky < 3; ++ky) {
            int iy = oy * 2 - 1 + ky;
            if (iy < 0 || iy >= IH) continue;
            for (int kx = 0; kx < 3; ++kx) {
                int ix = ox * 2 - 1 + kx;
                if (ix < 0 || ix >= IH) continue;
                acc += ip[iy * IH + ix] * wp[ky * 3 + kx];
            }
        }
    }
    out[idx] = acc > 0.0f ? acc : 0.0f;
}

// Last conv writes straight to f16 conv_tr[b][p][c]
__global__ void conv_s2_last(const float* __restrict__ in, const float* __restrict__ w,
                             const float* __restrict__ bias, _Float16* __restrict__ convF16) {
    int idx = blockIdx.x * THREADS + threadIdx.x;       // 32*24*8*8
    if (idx >= BATCH * CH * 64) return;
    const int IH = 16, OH = 8, IC = CH;
    int ox = idx % OH;
    int oy = (idx / OH) % OH;
    int oc = (idx / 64) % CH;
    int b  = idx / (CH * 64);
    float acc = bias[oc];
    const float* wc = w + oc * IC * 9;
    for (int ic = 0; ic < IC; ++ic) {
        const float* ip = in + ((size_t)(b * IC + ic) * IH) * IH;
        const float* wp = wc + ic * 9;
        for (int ky = 0; ky < 3; ++ky) {
            int iy = oy * 2 - 1 + ky;
            if (iy < 0 || iy >= IH) continue;
            for (int kx = 0; kx < 3; ++kx) {
                int ix = ox * 2 - 1 + kx;
                if (ix < 0 || ix >= IH) continue;
                acc += ip[iy * IH + ix] * wp[ky * 3 + kx];
            }
        }
    }
    acc = acc > 0.0f ? acc : 0.0f;
    int p = oy * 8 + ox;
    convF16[(b * NPAIR + p) * CC + oc] = (_Float16)acc;
}

// coords channels + lstm f16 pack
__global__ void pack_misc(const float* __restrict__ coords, const float* __restrict__ lstm,
                          _Float16* __restrict__ convF16, _Float16* __restrict__ lstmF16) {
    int tid = blockIdx.x * THREADS + threadIdx.x;       // 8192
    if (tid < BATCH * NPAIR * 2) {
        int b  = tid / (NPAIR * 2);
        int r  = tid % (NPAIR * 2);
        int p  = r % NPAIR;
        int cc = r / NPAIR;                              // 0 = yy, 1 = xx
        convF16[(b * NPAIR + p) * CC + 24 + cc] = (_Float16)coords[cc * 64 + p];
    } else if (tid < BATCH * NPAIR * 2 + BATCH * QDIM) {
        int t2 = tid - BATCH * NPAIR * 2;
        lstmF16[t2] = (_Float16)lstm[t2];                // [1,B,128] flat == [B,128]
    }
}

__global__ void zero_f32(float* __restrict__ p, int n) {
    int i = blockIdx.x * THREADS + threadIdx.x;
    if (i < n) p[i] = 0.0f;
}

// =====================================================================
// Templated per-layer compute: KT is compile-time so aF[] stays in VGPRs
// (no scratch spill) and all kt loops fully unroll into back-to-back
// ds_load_b128 + v_wmma with stationary A operands.
// =====================================================================
template <int KT, bool LAST>
__device__ __forceinline__ void gmlp_layer(const _Float16* __restrict__ aIn,
                                           _Float16* __restrict__ aOut,
                                           const _Float16* __restrict__ wbuf,
                                           const float* __restrict__ biasL,
                                           float* __restrict__ gsum,
                                           int mBase, int lane, int m, int hi) {
    // ---- A fragments (16x32 f16 layout): two 16B LDS loads each ----
    v16h aF[KT];
    {
        const _Float16* ar = aIn + (mBase + m) * AST + hi * 8;
#pragma unroll
        for (int kt = 0; kt < KT; ++kt) {
            union { v16h v; uint4 q[2]; } u;
            u.q[0] = *(const uint4*)(ar + kt * 32);
            u.q[1] = *(const uint4*)(ar + kt * 32 + 16);
            aF[kt] = u.v;
        }
    }

    // ---- 16 N-tiles, unrolled by 2 for independent WMMA chains ----
#pragma unroll 2
    for (int nt = 0; nt < 16; nt += 2) {
        v8f acc0 = {0.f, 0.f, 0.f, 0.f, 0.f, 0.f, 0.f, 0.f};
        v8f acc1 = {0.f, 0.f, 0.f, 0.f, 0.f, 0.f, 0.f, 0.f};
#pragma unroll
        for (int kt = 0; kt < KT; ++kt) {
            const _Float16* f0 = wbuf + (kt * 16 + nt) * 512 + lane * 8;
            const _Float16* f1 = f0 + 512;
            union { v16h v; uint4 q[2]; } u0, u1;
            u0.q[0] = *(const uint4*)f0;         u0.q[1] = *(const uint4*)(f0 + 256);
            u1.q[0] = *(const uint4*)f1;         u1.q[1] = *(const uint4*)(f1 + 256);
            acc0 = __builtin_amdgcn_wmma_f32_16x16x32_f16(
                false, aF[kt], false, u0.v, (short)0, acc0, false, false);
            acc1 = __builtin_amdgcn_wmma_f32_16x16x32_f16(
                false, aF[kt], false, u1.v, (short)0, acc1, false, false);
        }
        float bv0 = biasL[nt * 16 + m];
        float bv1 = biasL[(nt + 1) * 16 + m];
        if (!LAST) {
            // C/D layout: col = lane&15, row = reg + (hi?8:0)
            _Float16* o0 = aOut + (mBase + hi * 8) * AST + nt * 16 + m;
            _Float16* o1 = o0 + 16;
#pragma unroll
            for (int r = 0; r < 8; ++r) {
                float v0 = acc0[r] + bv0; v0 = v0 > 0.f ? v0 : 0.f;
                float v1 = acc1[r] + bv1; v1 = v1 > 0.f ? v1 : 0.f;
                o0[r * AST] = (_Float16)v0;
                o1[r * AST] = (_Float16)v1;
            }
        } else {
            // final layer: relu then reduce rows in-register, LDS-atomic per column
            float s0 = 0.f, s1 = 0.f;
#pragma unroll
            for (int r = 0; r < 8; ++r) {
                float v0 = acc0[r] + bv0; s0 += v0 > 0.f ? v0 : 0.f;
                float v1 = acc1[r] + bv1; s1 += v1 > 0.f ? v1 : 0.f;
            }
            atomicAdd(&gsum[nt * 16 + m], s0);
            atomicAdd(&gsum[(nt + 1) * 16 + m], s1);
        }
    }
}

// =====================================================================
// Fused relation core: pair construction + 4-layer g-MLP (WMMA f16) +
// sum over all pairs.  One WG = 128 pair rows of one batch, 8 waves x 16
// rows.  Weights + ping-pong activations live in LDS (262 KB of 320 KB).
// =====================================================================
__global__ void __launch_bounds__(THREADS, 1)
rn_gmlp(const _Float16* __restrict__ wfrag,
        const _Float16* __restrict__ convF16,
        const _Float16* __restrict__ lstmF16,
        const float* __restrict__ gb0, const float* __restrict__ gb1,
        const float* __restrict__ gb2, const float* __restrict__ gb3,
        float* __restrict__ gSumG) {
    extern __shared__ __align__(16) char smem[];
    _Float16* act0  = (_Float16*)smem;
    _Float16* act1  = act0 + SH_ACT;
    _Float16* wbuf  = act1 + SH_ACT;
    float*    biasL = (float*)(wbuf + SH_W);
    float*    gsum  = biasL + 256;

    const int t     = threadIdx.x;
    const int b     = blockIdx.x >> 4;
    const int chunk = blockIdx.x & 15;

    gsum[t] = 0.0f;

    // ---- build layer-0 activations: [convJ(26) | convI(26) | q(128) | pad] ----
    {
        int row  = t >> 1;
        int half = t & 1;
        int rb = chunk * 128 + row;                 // row within batch (0..2047)
        int i = rb >> 6, j = rb & 63;
        const _Float16* cj = convF16 + (b * NPAIR + j) * CC;
        const _Float16* ci = convF16 + (b * NPAIR + i) * CC;
        const _Float16* q  = lstmF16 + b * QDIM;
        _Float16* dst = act0 + row * AST;
        int c0 = half * 96, c1 = c0 + 96;
        for (int c = c0; c < c1; ++c) {
            _Float16 v;
            if      (c < CC)      v = cj[c];
            else if (c < 2 * CC)  v = ci[c - CC];
            else if (c < K0)      v = q[c - 2 * CC];
            else                  v = (_Float16)0.0f;
            dst[c] = v;
        }
    }

    const _Float16* wsrc[4] = {wfrag, wfrag + 49152, wfrag + 114688, wfrag + 180224};
    const float*    gbs[4]  = {gb0, gb1, gb2, gb3};

    const int wave  = t >> 5;
    const int lane  = t & 31;
    const int m     = lane & 15;
    const int hi    = lane >> 4;
    const int mBase = wave * 16;

    for (int L = 0; L < 4; ++L) {
        const int KT   = (L == 0) ? 6 : 8;         // K tiles of 32 (192 or 256)
        const int nvec = KT * 16 * 512 / 8;        // uint4 copies for weights
        __syncthreads();                           // prev layer done reading wbuf/act
        {
            const uint4* s = (const uint4*)wsrc[L];
            uint4* d = (uint4*)wbuf;
            for (int e = t; e < nvec; e += THREADS) d[e] = s[e];
            biasL[t] = gbs[L][t];
        }
        __syncthreads();

        const _Float16* aIn  = (L & 1) ? act1 : act0;
        _Float16*       aOut = (L & 1) ? act0 : act1;

        if (L == 0)
            gmlp_layer<6, false>(aIn, aOut, wbuf, biasL, gsum, mBase, lane, m, hi);
        else if (L < 3)
            gmlp_layer<8, false>(aIn, aOut, wbuf, biasL, gsum, mBase, lane, m, hi);
        else
            gmlp_layer<8, true>(aIn, aOut, wbuf, biasL, gsum, mBase, lane, m, hi);
    }

    __syncthreads();
    atomicAdd(&gSumG[b * 256 + t], gsum[t]);
}

// =====================================================================
// f-MLP head (tiny)
// =====================================================================
__global__ void f_mlp0(const float* __restrict__ gsum, const float* __restrict__ fw0,
                       const float* __restrict__ fb0, float* __restrict__ fhid) {
    int idx = blockIdx.x * THREADS + threadIdx.x;    // 32*256
    if (idx >= BATCH * HID) return;
    int oc = idx & 255, b = idx >> 8;
    const float* g = gsum + b * HID;
    const float* w = fw0 + oc * HID;
    float s = fb0[oc];
    for (int k = 0; k < HID; ++k) s += g[k] * w[k];
    fhid[idx] = s > 0.f ? s : 0.f;
}

__global__ void f_mlp1(const float* __restrict__ fhid, const float* __restrict__ fw1,
                       const float* __restrict__ fb1, float* __restrict__ out) {
    int idx = blockIdx.x * THREADS + threadIdx.x;    // 32*29
    if (idx >= BATCH * NANS) return;
    int a = idx % NANS, b = idx / NANS;
    const float* h = fhid + b * HID;
    const float* w = fw1 + a * HID;
    float s = fb1[a];
    for (int k = 0; k < HID; ++k) s += h[k] * w[k];
    out[idx] = s;
}

// =====================================================================
extern "C" void kernel_launch(void* const* d_in, const int* in_sizes, int n_in,
                              void* d_out, int out_size, void* d_ws, size_t ws_size,
                              hipStream_t stream) {
    (void)in_sizes; (void)n_in; (void)out_size; (void)ws_size;

    const float* image  = (const float*)d_in[0];
    const float* lstm   = (const float*)d_in[1];
    const float* coords = (const float*)d_in[2];
    const float* cw0 = (const float*)d_in[3],  *cb0 = (const float*)d_in[4];
    const float* cw1 = (const float*)d_in[5],  *cb1 = (const float*)d_in[6];
    const float* cw2 = (const float*)d_in[7],  *cb2 = (const float*)d_in[8];
    const float* cw3 = (const float*)d_in[9],  *cb3 = (const float*)d_in[10];
    const float* gw0 = (const float*)d_in[11], *gb0 = (const float*)d_in[12];
    const float* gw1 = (const float*)d_in[13], *gb1 = (const float*)d_in[14];
    const float* gw2 = (const float*)d_in[15], *gb2 = (const float*)d_in[16];
    const float* gw3 = (const float*)d_in[17], *gb3 = (const float*)d_in[18];
    const float* fw0 = (const float*)d_in[19], *fb0 = (const float*)d_in[20];
    const float* fw1 = (const float*)d_in[21], *fb1 = (const float*)d_in[22];
    float* out = (float*)d_out;

    char* ws = (char*)d_ws;
    _Float16* wf   = (_Float16*)(ws + OFF_W0);
    _Float16* conv = (_Float16*)(ws + OFF_CONV);
    _Float16* lst  = (_Float16*)(ws + OFF_LSTM);
    float* h0   = (float*)(ws + OFF_H0);
    float* h1   = (float*)(ws + OFF_H1);
    float* h2   = (float*)(ws + OFF_H2);
    float* gsum = (float*)(ws + OFF_GSUM);
    float* fhid = (float*)(ws + OFF_FHID);

    // weight repack to WMMA B-fragment order (f16)
    wprep<<<192, THREADS, 0, stream>>>(gw0, (_Float16*)(ws + OFF_W0), K0, 49152);
    wprep<<<256, THREADS, 0, stream>>>(gw1, (_Float16*)(ws + OFF_W1), HID, 65536);
    wprep<<<256, THREADS, 0, stream>>>(gw2, (_Float16*)(ws + OFF_W2), HID, 65536);
    wprep<<<256, THREADS, 0, stream>>>(gw3, (_Float16*)(ws + OFF_W3), HID, 65536);

    // conv stack
    conv_s2_relu<<<12288, THREADS, 0, stream>>>(image, cw0, cb0, h0, 3, 128, 64,
                                                BATCH * CH * 64 * 64);
    conv_s2_relu<<<3072, THREADS, 0, stream>>>(h0, cw1, cb1, h1, 24, 64, 32,
                                               BATCH * CH * 32 * 32);
    conv_s2_relu<<<768, THREADS, 0, stream>>>(h1, cw2, cb2, h2, 24, 32, 16,
                                              BATCH * CH * 16 * 16);
    conv_s2_last<<<192, THREADS, 0, stream>>>(h2, cw3, cb3, conv);
    pack_misc<<<32, THREADS, 0, stream>>>(coords, lstm, conv, lst);
    zero_f32<<<32, THREADS, 0, stream>>>(gsum, BATCH * HID);

    // fused relation core (WMMA)
    hipFuncSetAttribute(reinterpret_cast<const void*>(rn_gmlp),
                        hipFuncAttributeMaxDynamicSharedMemorySize, SMEM_BYTES);
    rn_gmlp<<<BATCH * 16, THREADS, SMEM_BYTES, stream>>>(wf, conv, lst,
                                                         gb0, gb1, gb2, gb3, gsum);

    // head
    f_mlp0<<<32, THREADS, 0, stream>>>(gsum, fw0, fb0, fhid);
    f_mlp1<<<4, THREADS, 0, stream>>>(fhid, fw1, fb1, out);
}
